// FastRCNNOutputLayersIoU_64012192579930
// MI455X (gfx1250) — compile-verified, matrix operand order checked by reading, not run on hardware
//
#include <hip/hip_runtime.h>

typedef __attribute__((ext_vector_type(4)))  float  v4f;
typedef __attribute__((ext_vector_type(8)))  float  v8f;
typedef __attribute__((ext_vector_type(4)))  __bf16 v4bf;
typedef __attribute__((ext_vector_type(8)))  __bf16 v8bf;
typedef __attribute__((ext_vector_type(16))) __bf16 v16bf;

#define ROWS   65536
#define KD     1024
#define NCLS   81          // NUM_CLASSES + 1
#define NBOX   320         // NUM_CLASSES * 4
#define NTOT   482         // 81 + 320 + 81
#define NPAD   496         // 482 padded to 31 tiles of 16 (zero columns 482..495)
#define NTILES 31
#define BM     64          // rows per workgroup (LDS-limited)
#define LDH    1032        // LDS row stride in bf16: 1024 + 8 -> b128 frags hit distinct banks
#define XLO    (BM * LDH)  // offset of lo plane in LDS (bf16 elements)

// ---------------------------------------------------------------------------
// Pre-pass: pack W = [W_cls | W_bbox | W_iou] (fp32, row-major, 3 arrays) into
// d_ws as column-major bf16 hi/lo planes [NPAD][KD]. Region mapping + zero
// padding resolved here once, so the GEMM hot loop has zero branching.
// ---------------------------------------------------------------------------
extern "C" __global__ void __launch_bounds__(256)
frcnn_pack_w(const float* __restrict__ Wc, const float* __restrict__ Wb,
             const float* __restrict__ Wi, __bf16* __restrict__ wp)
{
    const int idx = blockIdx.x * 256 + threadIdx.x;   // NPAD*KD total
    const int k = idx & (KD - 1);                     // consecutive threads -> consecutive k (coalesced stores)
    const int n = idx >> 10;
    float v;
    if (n < NCLS)              v = Wc[(size_t)k * NCLS + n];
    else if (n < NCLS + NBOX)  v = Wb[(size_t)k * NBOX + (n - NCLS)];
    else if (n < NTOT)         v = Wi[(size_t)k * NCLS + (n - NCLS - NBOX)];
    else                       v = 0.f;
    const __bf16 h = (__bf16)v;
    const __bf16 l = (__bf16)(v - (float)h);          // residual: hi+lo ~ 16 mantissa bits
    wp[(size_t)n * KD + k]                         = h;   // hi plane, column-major
    wp[(size_t)NPAD * KD + (size_t)n * KD + k]     = l;   // lo plane, column-major
}

// ---------------------------------------------------------------------------
// Main GEMM: 3xBF16 split (ah*bh + ah*bl + al*bh) with v_wmma_f32_16x16x32_bf16
// accumulating in fp32 -> ~fp32 accuracy at the bf16 matrix-pipe rate.
// One workgroup = 8 waves: 4 row-subtiles x 2 column-tile parities over a
// 64-row x-stripe staged once into LDS (bf16 hi/lo planes, 264KB of 320KB).
// ---------------------------------------------------------------------------
extern "C" __global__ void __launch_bounds__(256)
frcnn_head_wmma(const float* __restrict__ x,
                const float* __restrict__ bc, const float* __restrict__ bb,
                const float* __restrict__ bi,
                const __bf16* __restrict__ wp,
                float* __restrict__ out)
{
    extern __shared__ __bf16 xs[];                    // [2][BM][LDH] hi then lo
    const int tid  = threadIdx.x;
    const int lane = tid & 31;
    const int wave = tid >> 5;
    const int row0 = blockIdx.x * BM;

    // ---- Stage + split-convert the 64x1024 x-block (x read once from HBM) ----
    {
        const v4f* xg = (const v4f*)(x + (size_t)row0 * KD);
        #pragma unroll 4
        for (int r = 0; r < BM; ++r) {
            v4f v = __builtin_nontemporal_load(&xg[r * (KD / 4) + tid]); // row r, coalesced
            v4bf h, l;
            #pragma unroll
            for (int j = 0; j < 4; ++j) {
                const __bf16 hh = (__bf16)v[j];
                h[j] = hh;
                l[j] = (__bf16)(v[j] - (float)hh);
            }
            *(v4bf*)(&xs[r * LDH + tid * 4])       = h;
            *(v4bf*)(&xs[XLO + r * LDH + tid * 4]) = l;
        }
    }
    __syncthreads();

    const int row_sub = wave & 3;                     // 16-row subtile
    const int tpar    = wave >> 2;                    // column-tile parity
    const int nlane   = lane & 15;                    // N index / A row (M)
    const int half    = lane >> 4;                    // K-half select per ISA fragment layout

    // A fragment base: lane holds M = row, K = k0+8h..+7 (V0-3) and +16 (V4-7)
    const __bf16* ah = xs + (row_sub * 16 + nlane) * LDH + 8 * half;
    const __bf16* al = ah + XLO;

    for (int tile = tpar; tile < NTILES; tile += 2) {
        const int col = tile * 16 + nlane;            // 0..495: always valid in packed W
        const __bf16* bh = wp + (size_t)col * KD + 8 * half;   // column-major packed W
        const __bf16* bl = bh + (size_t)NPAD * KD;

        v8f c = {0.f, 0.f, 0.f, 0.f, 0.f, 0.f, 0.f, 0.f};

        #pragma unroll 4
        for (int k0 = 0; k0 < KD; k0 += 32) {
            // Each operand = two contiguous 16B pieces: K(k0+8h..) and K(k0+16+8h..)
            v8bf a0 = *(const v8bf*)(ah + k0);
            v8bf a1 = *(const v8bf*)(ah + k0 + 16);
            v8bf c0 = *(const v8bf*)(al + k0);
            v8bf c1 = *(const v8bf*)(al + k0 + 16);
            v8bf b0 = *(const v8bf*)(bh + k0);
            v8bf b1 = *(const v8bf*)(bh + k0 + 16);
            v8bf d0 = *(const v8bf*)(bl + k0);
            v8bf d1 = *(const v8bf*)(bl + k0 + 16);
            v16bf Ah = __builtin_shufflevector(a0, a1, 0,1,2,3,4,5,6,7,8,9,10,11,12,13,14,15);
            v16bf Al = __builtin_shufflevector(c0, c1, 0,1,2,3,4,5,6,7,8,9,10,11,12,13,14,15);
            v16bf Bh = __builtin_shufflevector(b0, b1, 0,1,2,3,4,5,6,7,8,9,10,11,12,13,14,15);
            v16bf Bl = __builtin_shufflevector(d0, d1, 0,1,2,3,4,5,6,7,8,9,10,11,12,13,14,15);
            // 3xBF16 split: drop only the (al*bl) term (~2^-16 relative)
            c = __builtin_amdgcn_wmma_f32_16x16x32_bf16(false, Ah, false, Bh, (short)0, c, false, false);
            c = __builtin_amdgcn_wmma_f32_16x16x32_bf16(false, Ah, false, Bl, (short)0, c, false, false);
            c = __builtin_amdgcn_wmma_f32_16x16x32_bf16(false, Al, false, Bh, (short)0, c, false, false);
        }

        // ---- bias + scatter into the three flattened outputs ----
        const bool valid = (col < NTOT);
        const int  jc    = valid ? col : 0;
        float bias;
        if (jc < NCLS)             bias = bc[jc];
        else if (jc < NCLS + NBOX) bias = bb[jc - NCLS];
        else                       bias = bi[jc - NCLS - NBOX];

        if (valid) {
            #pragma unroll
            for (int i = 0; i < 8; ++i) {             // VGPR i: M = i (lanes 0-15) / i+8 (16-31)
                const int   grow = row0 + row_sub * 16 + i + half * 8;
                const float val  = c[i] + bias;
                size_t off;
                if (col < NCLS)
                    off = (size_t)grow * NCLS + col;                                              // scores
                else if (col < NCLS + NBOX)
                    off = (size_t)ROWS * NCLS + (size_t)grow * NBOX + (col - NCLS);               // deltas
                else
                    off = (size_t)ROWS * (NCLS + NBOX) + (size_t)grow * NCLS + (col - NCLS - NBOX); // ious
                __builtin_nontemporal_store(val, &out[off]);                                      // streamed
            }
        }
    }
}

extern "C" void kernel_launch(void* const* d_in, const int* in_sizes, int n_in,
                              void* d_out, int out_size, void* d_ws, size_t ws_size,
                              hipStream_t stream) {
    const float* x  = (const float*)d_in[0];
    const float* Wc = (const float*)d_in[1];
    const float* bc = (const float*)d_in[2];
    const float* Wb = (const float*)d_in[3];
    const float* bb = (const float*)d_in[4];
    const float* Wi = (const float*)d_in[5];
    const float* bi = (const float*)d_in[6];
    float*  out = (float*)d_out;
    __bf16* wp  = (__bf16*)d_ws;                      // needs 2*NPAD*KD*2B ~ 1.94 MB scratch

    // Pass 1: pack W into column-major bf16 hi/lo planes (region logic resolved once)
    frcnn_pack_w<<<dim3((NPAD * KD) / 256), dim3(256), 0, stream>>>(Wc, Wb, Wi, wp);

    // Pass 2: split-bf16 WMMA GEMM
    const size_t shmem = (size_t)2 * BM * LDH * sizeof(__bf16);   // 264192 B < 320 KB WGP LDS
    (void)hipFuncSetAttribute((const void*)frcnn_head_wmma,
                              hipFuncAttributeMaxDynamicSharedMemorySize, (int)shmem);
    frcnn_head_wmma<<<dim3(ROWS / BM), dim3(256), shmem, stream>>>(x, bc, bb, bi, wp, out);
}
